// AttentionGRUCell_10617159155943
// MI455X (gfx1250) — matrix operand order repaired and checked by reading, not compiled, not run on hardware
//
#include <hip/hip_runtime.h>
#include <hip/hip_bf16.h>
#include <math.h>

#define B_  128
#define T_  256
#define IN_ 1024
#define H_  1024
#define NE_ 512
#define KI_ (IN_ + NE_)   // 1536

typedef __attribute__((ext_vector_type(16))) __bf16 v16bf;
typedef __attribute__((ext_vector_type(8)))  float  v8f;
typedef __attribute__((ext_vector_type(4)))  float  f32x4;
typedef __attribute__((ext_vector_type(2)))  unsigned int u32x2;

// ---------- helpers ----------
__device__ __forceinline__ unsigned short f2bf(float f) {
  unsigned int u = __float_as_uint(f);
  u += 0x7FFFu + ((u >> 16) & 1u);   // round-to-nearest-even
  return (unsigned short)(u >> 16);
}

// CDNA5 async copy: global -> LDS, 16 bytes per lane, tracked by ASYNCcnt.
__device__ __forceinline__ void async_b128_to_lds(unsigned short* lds_ptr,
                                                  const unsigned short* gptr) {
  unsigned loff = (unsigned)(uintptr_t)lds_ptr;  // low 32 bits of generic LDS addr = LDS offset
  asm volatile("global_load_async_to_lds_b128 %0, %1, off"
               :: "v"(loff), "v"((unsigned long long)(uintptr_t)gptr)
               : "memory");
}
__device__ __forceinline__ void wait_async_zero() {
  asm volatile("s_wait_asynccnt 0x0" ::: "memory");
}

// ---------- 1) fp32 -> bf16 conversion (grid over float4s) ----------
__global__ void cvt_bf16_kernel(const float* __restrict__ src,
                                unsigned short* __restrict__ dst, int n4) {
  int i = blockIdx.x * blockDim.x + threadIdx.x;
  if (i >= n4) return;
  f32x4 v = ((const f32x4*)src)[i];
  u32x2 p;
  p.x = (unsigned int)f2bf(v.x) | ((unsigned int)f2bf(v.y) << 16);
  p.y = (unsigned int)f2bf(v.z) | ((unsigned int)f2bf(v.w) << 16);
  ((u32x2*)dst)[i] = p;
}

// strided variant: pack [rows x cols] fp32 into dst rows of dst_stride at dst_off (elems)
__global__ void cvt_bf16_strided_kernel(const float* __restrict__ src,
                                        unsigned short* __restrict__ dst,
                                        int cols, int dst_stride, int dst_off) {
  int idx = blockIdx.x * blockDim.x + threadIdx.x;   // rows*cols/4
  int c4n = cols >> 2;
  int r = idx / c4n, c4 = idx % c4n;
  f32x4 v = ((const f32x4*)(src + (size_t)r * cols))[c4];
  u32x2 p;
  p.x = (unsigned int)f2bf(v.x) | ((unsigned int)f2bf(v.y) << 16);
  p.y = (unsigned int)f2bf(v.z) | ((unsigned int)f2bf(v.w) << 16);
  ((u32x2*)(dst + (size_t)r * dst_stride + dst_off))[c4] = p;
}

// ---------- WMMA tiling config ----------
#define MT 64
#define NT 128
#define KC 128
#define APITCH 136   // bf16 elems per LDS row (272B, staggers banks)
#define BPITCH 136

union AF { v16bf v; f32x4 f[2]; };

// ---------- 2) fused WMMA: e[m] = sum_n tanh(A@B^T + hproj) * w_score ----------
__global__ __launch_bounds__(256) void attn_score_kernel(
    const unsigned short* __restrict__ Abf,   // [B*T][IN] bf16 (batch_H)
    const unsigned short* __restrict__ Wbf,   // [H][IN]  bf16 (W_i2h)
    const float* __restrict__ hproj,          // [B][H]
    const float* __restrict__ w_score,        // [H]
    float* __restrict__ e_out) {              // [B*T]
  __shared__ unsigned short Alds[MT * APITCH];
  __shared__ unsigned short Blds[NT * BPITCH];
  __shared__ float esum[2][MT];

  const int tid    = threadIdx.x;
  const int wave   = tid >> 5;
  const int lane   = tid & 31;
  const int lane16 = lane & 15;
  const int sel    = lane >> 4;
  const int m_sub  = wave & 3;
  const int n_grp  = wave >> 2;

  const int m0   = blockIdx.x * MT;
  const int bidx = m0 >> 8;

  float rowAcc[8];
#pragma unroll
  for (int j = 0; j < 8; ++j) rowAcc[j] = 0.0f;

  for (int n0 = 0; n0 < H_; n0 += NT) {
    v8f c[4];
#pragma unroll
    for (int ns = 0; ns < 4; ++ns)
#pragma unroll
      for (int j = 0; j < 8; ++j) c[ns][j] = 0.0f;

    for (int k0 = 0; k0 < IN_; k0 += KC) {
      __syncthreads();
      for (int i = tid; i < MT * (KC / 8); i += 256) {
        int r = i >> 4, cc = i & 15;
        async_b128_to_lds(&Alds[r * APITCH + cc * 8],
                          &Abf[(size_t)(m0 + r) * IN_ + k0 + cc * 8]);
      }
      for (int i = tid; i < NT * (KC / 8); i += 256) {
        int r = i >> 4, cc = i & 15;
        async_b128_to_lds(&Blds[r * BPITCH + cc * 8],
                          &Wbf[(size_t)(n0 + r) * IN_ + k0 + cc * 8]);
      }
      wait_async_zero();
      __syncthreads();

#pragma unroll
      for (int kk = 0; kk < KC; kk += 32) {
        AF a;
        const int arow = m_sub * 16 + lane16;
        a.f[0] = *(const f32x4*)&Alds[arow * APITCH + kk + sel * 8];
        a.f[1] = *(const f32x4*)&Alds[arow * APITCH + kk + 16 + sel * 8];
#pragma unroll
        for (int ns = 0; ns < 4; ++ns) {
          AF bf;
          const int brow = n_grp * 64 + ns * 16 + lane16;
          bf.f[0] = *(const f32x4*)&Blds[brow * BPITCH + kk + sel * 8];
          bf.f[1] = *(const f32x4*)&Blds[brow * BPITCH + kk + 16 + sel * 8];
          c[ns] = __builtin_amdgcn_wmma_f32_16x16x32_bf16(
              false, a.v, false, bf.v, (short)0, c[ns], false, false);
        }
      }
    }

    // fused epilogue: tanh(c + hproj[b,n]) * w_score[n], accumulate per row
#pragma unroll
    for (int ns = 0; ns < 4; ++ns) {
      const int ncol = n0 + n_grp * 64 + ns * 16 + lane16;
      const float hp  = hproj[(size_t)bidx * H_ + ncol];
      const float wsc = w_score[ncol];
#pragma unroll
      for (int j = 0; j < 8; ++j)
        rowAcc[j] += tanhf(c[ns][j] + hp) * wsc;
    }
  }

#pragma unroll
  for (int mask = 1; mask < 16; mask <<= 1)
#pragma unroll
    for (int j = 0; j < 8; ++j)
      rowAcc[j] += __shfl_xor(rowAcc[j], mask, 32);

  if (lane16 == 0) {
#pragma unroll
    for (int j = 0; j < 8; ++j)
      esum[n_grp][m_sub * 16 + sel * 8 + j] = rowAcc[j];
  }
  __syncthreads();
  if (tid < MT) e_out[m0 + tid] = esum[0][tid] + esum[1][tid];
}

// ---------- 3) generic WMMA GEMM: C[M,N] = A[M,K] @ B[N,K]^T + bias[N] ----------
// grid: (N/NT, M/MT), block 256. K multiple of 128.
__global__ __launch_bounds__(256) void wmma_gemm_bias_kernel(
    const unsigned short* __restrict__ Abf,   // [M][K] bf16
    const unsigned short* __restrict__ Bbf,   // [N][K] bf16
    const float* __restrict__ bias,           // [N]
    float* __restrict__ C,                    // [M][N] fp32
    int K, int N) {
  __shared__ unsigned short Alds[MT * APITCH];
  __shared__ unsigned short Blds[NT * BPITCH];

  const int tid    = threadIdx.x;
  const int wave   = tid >> 5;
  const int lane   = tid & 31;
  const int lane16 = lane & 15;
  const int sel    = lane >> 4;
  const int m_sub  = wave & 3;
  const int n_grp  = wave >> 2;

  const int n0 = blockIdx.x * NT;
  const int m0 = blockIdx.y * MT;

  v8f c[4];
#pragma unroll
  for (int ns = 0; ns < 4; ++ns)
#pragma unroll
    for (int j = 0; j < 8; ++j) c[ns][j] = 0.0f;

  for (int k0 = 0; k0 < K; k0 += KC) {
    __syncthreads();
    for (int i = tid; i < MT * (KC / 8); i += 256) {
      int r = i >> 4, cc = i & 15;
      async_b128_to_lds(&Alds[r * APITCH + cc * 8],
                        &Abf[(size_t)(m0 + r) * (size_t)K + k0 + cc * 8]);
    }
    for (int i = tid; i < NT * (KC / 8); i += 256) {
      int r = i >> 4, cc = i & 15;
      async_b128_to_lds(&Blds[r * BPITCH + cc * 8],
                        &Bbf[(size_t)(n0 + r) * (size_t)K + k0 + cc * 8]);
    }
    wait_async_zero();
    __syncthreads();

#pragma unroll
    for (int kk = 0; kk < KC; kk += 32) {
      AF a;
      const int arow = m_sub * 16 + lane16;
      a.f[0] = *(const f32x4*)&Alds[arow * APITCH + kk + sel * 8];
      a.f[1] = *(const f32x4*)&Alds[arow * APITCH + kk + 16 + sel * 8];
#pragma unroll
      for (int ns = 0; ns < 4; ++ns) {
        AF bf;
        const int brow = n_grp * 64 + ns * 16 + lane16;
        bf.f[0] = *(const f32x4*)&Blds[brow * BPITCH + kk + sel * 8];
        bf.f[1] = *(const f32x4*)&Blds[brow * BPITCH + kk + 16 + sel * 8];
        c[ns] = __builtin_amdgcn_wmma_f32_16x16x32_bf16(
            false, a.v, false, bf.v, (short)0, c[ns], false, false);
      }
    }
  }

#pragma unroll
  for (int ns = 0; ns < 4; ++ns) {
    const int n = n0 + n_grp * 64 + ns * 16 + lane16;
    const float bv = bias ? bias[n] : 0.0f;
#pragma unroll
    for (int j = 0; j < 8; ++j) {
      const int m = m0 + m_sub * 16 + sel * 8 + j;
      C[(size_t)m * N + n] = c[ns][j] + bv;
    }
  }
}

// ---------- 4) softmax over T per batch ----------
__global__ void softmax_kernel(const float* __restrict__ e,
                               float* __restrict__ alpha) {
  __shared__ float red[T_];
  int b = blockIdx.x, t = threadIdx.x;
  float v = e[(size_t)b * T_ + t];
  red[t] = v;
  __syncthreads();
  for (int s = T_ / 2; s > 0; s >>= 1) {
    if (t < s) red[t] = fmaxf(red[t], red[t + s]);
    __syncthreads();
  }
  float m = red[0];
  __syncthreads();
  float ex = expf(v - m);
  red[t] = ex;
  __syncthreads();
  for (int s = T_ / 2; s > 0; s >>= 1) {
    if (t < s) red[t] += red[t + s];
    __syncthreads();
  }
  alpha[(size_t)b * T_ + t] = ex / red[0];
}

// ---------- 5) context[b,i] = sum_t alpha[b,t] * batch_H[b,t,i] ----------
__global__ void context_kernel(const float* __restrict__ alpha,
                               const float* __restrict__ batchH,
                               float* __restrict__ ctx) {
  int idx = blockIdx.x * blockDim.x + threadIdx.x;   // B*IN
  int b = idx >> 10, i = idx & (IN_ - 1);
  const float* bh = batchH + (size_t)b * T_ * IN_ + i;
  const float* al = alpha + (size_t)b * T_;
  float acc = 0.0f;
#pragma unroll 4
  for (int t = 0; t < T_; ++t) acc += al[t] * bh[(size_t)t * IN_];
  ctx[idx] = acc;
}

// ---------- 6) GRU gates from precomputed gi, gh ----------
__global__ void gate_kernel(const float* __restrict__ gi,
                            const float* __restrict__ gh,
                            const float* __restrict__ prev,
                            float* __restrict__ out) {
  int idx = blockIdx.x * blockDim.x + threadIdx.x;   // B*H
  int b = idx >> 10, h = idx & (H_ - 1);
  const float* gib = gi + (size_t)b * 3 * H_;
  const float* ghb = gh + (size_t)b * 3 * H_;
  float gir = gib[h], giz = gib[H_ + h], gin = gib[2 * H_ + h];
  float ghr = ghb[h], ghz = ghb[H_ + h], ghn = ghb[2 * H_ + h];
  float r = 1.0f / (1.0f + expf(-(gir + ghr)));
  float z = 1.0f / (1.0f + expf(-(giz + ghz)));
  float n = tanhf(gin + r * ghn);
  out[idx] = (1.0f - z) * n + z * prev[idx];
}

// ---------- host ----------
extern "C" void kernel_launch(void* const* d_in, const int* in_sizes, int n_in,
                              void* d_out, int out_size, void* d_ws, size_t ws_size,
                              hipStream_t stream) {
  const float* prev    = (const float*)d_in[0];
  const float* batchH  = (const float*)d_in[1];
  const float* onehot  = (const float*)d_in[2];
  const float* W_i2h   = (const float*)d_in[3];
  const float* W_h2h   = (const float*)d_in[4];
  const float* b_h2h   = (const float*)d_in[5];
  const float* W_score = (const float*)d_in[6];
  const float* W_ih    = (const float*)d_in[7];
  const float* W_hh    = (const float*)d_in[8];
  const float* b_ih    = (const float*)d_in[9];
  const float* b_hh    = (const float*)d_in[10];

  float* out        = (float*)d_out;
  float* cur_hidden = out;                       // [B,H]
  float* alpha      = out + (size_t)B_ * H_;     // [B,1,T]

  char* ws = (char*)d_ws;
  size_t off = 0;
  unsigned short* bhb   = (unsigned short*)(ws + off); off += (size_t)B_ * T_ * IN_ * 2;
  unsigned short* wib   = (unsigned short*)(ws + off); off += (size_t)H_ * IN_ * 2;
  unsigned short* prevb = (unsigned short*)(ws + off); off += (size_t)B_ * H_ * 2;
  unsigned short* wh2hb = (unsigned short*)(ws + off); off += (size_t)H_ * H_ * 2;
  unsigned short* wihb  = (unsigned short*)(ws + off); off += (size_t)3 * H_ * KI_ * 2;
  unsigned short* whhb  = (unsigned short*)(ws + off); off += (size_t)3 * H_ * H_ * 2;
  unsigned short* xb    = (unsigned short*)(ws + off); off += (size_t)B_ * KI_ * 2;
  float* hproj = (float*)(ws + off); off += (size_t)B_ * H_ * 4;
  float* evec  = (float*)(ws + off); off += (size_t)B_ * T_ * 4;
  float* ctx   = (float*)(ws + off); off += (size_t)B_ * IN_ * 4;
  float* gi    = (float*)(ws + off); off += (size_t)B_ * 3 * H_ * 4;
  float* gh    = (float*)(ws + off); off += (size_t)B_ * 3 * H_ * 4;
  (void)off; (void)ws_size; (void)in_sizes; (void)n_in; (void)out_size;

  // bf16 conversions
  cvt_bf16_kernel<<<(B_ * T_ * IN_ / 4) / 256, 256, 0, stream>>>(batchH, bhb, B_ * T_ * IN_ / 4);
  cvt_bf16_kernel<<<(H_ * IN_ / 4) / 256, 256, 0, stream>>>(W_i2h, wib, H_ * IN_ / 4);
  cvt_bf16_kernel<<<(B_ * H_ / 4) / 256, 256, 0, stream>>>(prev, prevb, B_ * H_ / 4);
  cvt_bf16_kernel<<<(H_ * H_ / 4) / 256, 256, 0, stream>>>(W_h2h, wh2hb, H_ * H_ / 4);
  cvt_bf16_kernel<<<(3 * H_ * KI_ / 4) / 256, 256, 0, stream>>>(W_ih, wihb, 3 * H_ * KI_ / 4);
  cvt_bf16_kernel<<<(3 * H_ * H_ / 4) / 256, 256, 0, stream>>>(W_hh, whhb, 3 * H_ * H_ / 4);

  // hproj = prev @ W_h2h^T + b_h2h   (WMMA)
  wmma_gemm_bias_kernel<<<dim3(H_ / NT, B_ / MT), 256, 0, stream>>>(
      prevb, wh2hb, b_h2h, hproj, H_, H_);

  // attention scores (WMMA, fused tanh + score-dot)
  attn_score_kernel<<<(B_ * T_) / MT, 256, 0, stream>>>(bhb, wib, hproj, W_score, evec);

  // softmax -> alpha (directly into output slot)
  softmax_kernel<<<B_, T_, 0, stream>>>(evec, alpha);

  // context = alpha^T @ batch_H (fp32, L2-resident)
  context_kernel<<<(B_ * IN_) / 256, 256, 0, stream>>>(alpha, batchH, ctx);

  // x = [context | onehots] -> bf16, row stride KI_
  cvt_bf16_strided_kernel<<<(B_ * IN_ / 4) / 256, 256, 0, stream>>>(ctx, xb, IN_, KI_, 0);
  cvt_bf16_strided_kernel<<<(B_ * NE_ / 4) / 256, 256, 0, stream>>>(onehot, xb, NE_, KI_, IN_);

  // gi = x @ W_ih^T + b_ih ; gh = prev @ W_hh^T + b_hh   (WMMA)
  wmma_gemm_bias_kernel<<<dim3(3 * H_ / NT, B_ / MT), 256, 0, stream>>>(
      xb, wihb, b_ih, gi, KI_, 3 * H_);
  wmma_gemm_bias_kernel<<<dim3(3 * H_ / NT, B_ / MT), 256, 0, stream>>>(
      prevb, whhb, b_hh, gh, H_, 3 * H_);

  // GRU gates
  gate_kernel<<<(B_ * H_) / 256, 256, 0, stream>>>(gi, gh, prev, cur_hidden);
}
